// Model_13486197309850
// MI455X (gfx1250) — matrix operand order repaired and checked by reading, not compiled, not run on hardware
//
#include <hip/hip_runtime.h>
#include <hip/hip_bf16.h>
#include <math.h>

typedef __bf16 bf16;
typedef __bf16 v16bf __attribute__((ext_vector_type(16)));
typedef __bf16 v8bf  __attribute__((ext_vector_type(8)));
typedef float  v8f   __attribute__((ext_vector_type(8)));

#define WMMA_BF16(A_, B_, C_) \
  __builtin_amdgcn_wmma_f32_16x16x32_bf16(false, (A_), false, (B_), (short)0, (C_), false, false)

// CDNA5 async global->LDS DMA (GLOBAL_LOAD_ASYNC_TO_LDS_B128, ASYNCcnt-tracked).
#if defined(__HIP_DEVICE_COMPILE__) && defined(__has_builtin)
#if __has_builtin(__builtin_amdgcn_global_load_async_to_lds_b128) && \
    __has_builtin(__builtin_amdgcn_s_wait_asynccnt)
#define ASYNC_LDS 1
#endif
#endif

#ifdef ASYNC_LDS
typedef int v4i_vec __attribute__((vector_size(16)));
typedef __attribute__((address_space(1))) v4i_vec gv4i_t;  // global int4*
typedef __attribute__((address_space(3))) v4i_vec lv4i_t;  // LDS int4*
__device__ __forceinline__ void async_b128(const bf16* gsrc, bf16* ldst) {
  __builtin_amdgcn_global_load_async_to_lds_b128(
      (gv4i_t*)(__attribute__((address_space(1))) void*)(void*)gsrc,
      (lv4i_t*)(__attribute__((address_space(3))) void*)(void*)ldst, 0, 0);
}
__device__ __forceinline__ void wait_async0() { __builtin_amdgcn_s_wait_asynccnt(0); }
#endif

// ---------------------------------------------------------------------------
// Elementwise helpers
// ---------------------------------------------------------------------------
__global__ void f2bf_kernel(const float* __restrict__ in, bf16* __restrict__ out, size_t n) {
  size_t i = (size_t)blockIdx.x * blockDim.x + threadIdx.x;
  if (i < n) out[i] = (bf16)in[i];
}

// in (R,C) row-major -> out (C,R) row-major (bf16)
__global__ void transpose_f2bf_kernel(const float* __restrict__ in, bf16* __restrict__ out,
                                      int R, int C) {
  size_t i = (size_t)blockIdx.x * blockDim.x + threadIdx.x;
  if (i < (size_t)R * C) {
    int r = (int)(i / C), c = (int)(i % C);
    out[(size_t)c * R + r] = (bf16)in[i];
  }
}

__global__ void zero_f32_kernel(float* __restrict__ p, size_t n) {
  size_t i = (size_t)blockIdx.x * blockDim.x + threadIdx.x;
  if (i < n) p[i] = 0.0f;
}

// ---------------------------------------------------------------------------
// bf16 WMMA GEMM: C[M,N] = A[M,K] * B  (B stored n-major: [N][K]).
// Block tile 128x256, BK=32, double-buffered LDS, 8 waves, 16 WMMA/wave/K-step.
// M % 128 == 0, N % 256 == 0, K % 32 == 0.
// act: 0 = none, 1 = relu, 2 = leaky_relu(0.01)
// ---------------------------------------------------------------------------
__global__ __launch_bounds__(256) void gemm_bf16_kernel(
    const bf16* __restrict__ A, const bf16* __restrict__ Bn, float* __restrict__ C,
    int M, int N, int K, const float* __restrict__ bias, int act) {
  __shared__ bf16 la[2][128 * 32];
  __shared__ bf16 lb[2][256 * 32];

  const int t     = threadIdx.x;
  const int lane  = t & 31;
  const int w     = t >> 5;
  const int wm    = w & 3;   // 4 row-groups of 32
  const int wn    = w >> 2;  // 2 col-groups of 128
  const int bm0   = blockIdx.y * 128;
  const int bn0   = blockIdx.x * 256;
  const int mrow  = lane & 15;
  const int khalf = lane >> 4;

  const int arow = t >> 1, ahalf = t & 1;  // A tile: 2 threads per row (32B each)
  const int brow = t;                      // B tile: 1 thread per n-row (64B)

  const bf16* aptr = A + (size_t)(bm0 + arow) * K + ahalf * 16;
  const bf16* bptr = Bn + (size_t)(bn0 + brow) * K;
  bf16* la_dst0 = &la[0][arow * 32 + ahalf * 16];
  bf16* lb_dst0 = &lb[0][brow * 32];
  bf16* la_dst1 = &la[1][arow * 32 + ahalf * 16];
  bf16* lb_dst1 = &lb[1][brow * 32];

  v8f acc[2][8];
#pragma unroll
  for (int mi = 0; mi < 2; ++mi)
#pragma unroll
    for (int ni = 0; ni < 8; ++ni)
#pragma unroll
      for (int e = 0; e < 8; ++e) acc[mi][ni][e] = 0.0f;

  // ---- stage tile 0 into buffer 0 ----
#ifdef ASYNC_LDS
  async_b128(aptr, la_dst0);
  async_b128(aptr + 8, la_dst0 + 8);
#pragma unroll
  for (int kq = 0; kq < 4; ++kq) async_b128(bptr + kq * 8, lb_dst0 + kq * 8);
  wait_async0();
#else
  *(v8bf*)la_dst0       = *(const v8bf*)(aptr);
  *(v8bf*)(la_dst0 + 8) = *(const v8bf*)(aptr + 8);
#pragma unroll
  for (int kq = 0; kq < 4; ++kq)
    *(v8bf*)(lb_dst0 + kq * 8) = *(const v8bf*)(bptr + kq * 8);
#endif
  __syncthreads();

  int cur = 0;
  for (int kk = 0; kk < K; kk += 32) {
    const int nxt = cur ^ 1;
    const bool have_next = (kk + 32) < K;
    bf16* la_nxt = nxt ? la_dst1 : la_dst0;
    bf16* lb_nxt = nxt ? lb_dst1 : lb_dst0;

    // ---- start fetching tile kk+32 into the other buffer ----
#ifdef ASYNC_LDS
    if (have_next) {
      async_b128(aptr + kk + 32, la_nxt);
      async_b128(aptr + kk + 40, la_nxt + 8);
#pragma unroll
      for (int kq = 0; kq < 4; ++kq)
        async_b128(bptr + kk + 32 + kq * 8, lb_nxt + kq * 8);
    }
#else
    v8bf ra0, ra1, rb[4];
    if (have_next) {
      ra0 = *(const v8bf*)(aptr + kk + 32);
      ra1 = *(const v8bf*)(aptr + kk + 40);
#pragma unroll
      for (int kq = 0; kq < 4; ++kq) rb[kq] = *(const v8bf*)(bptr + kk + 32 + kq * 8);
      __builtin_prefetch(bptr + kk + 64, 0, 1);
    }
#endif

    // ---- compute on current buffer ----
    const bf16* lac = la[cur];
    const bf16* lbc = lb[cur];
    v16bf afrag[2];
#pragma unroll
    for (int mi = 0; mi < 2; ++mi) {
      const int m0 = wm * 32 + mi * 16;
      v8bf a0 = *(const v8bf*)&lac[(m0 + mrow) * 32 + khalf * 8];
      v8bf a1 = *(const v8bf*)&lac[(m0 + mrow) * 32 + 16 + khalf * 8];
#pragma unroll
      for (int e = 0; e < 8; ++e) { afrag[mi][e] = a0[e]; afrag[mi][e + 8] = a1[e]; }
    }
#pragma unroll
    for (int ni = 0; ni < 8; ++ni) {
      const int n0 = wn * 128 + ni * 16;
      v8bf b0 = *(const v8bf*)&lbc[(n0 + mrow) * 32 + khalf * 8];
      v8bf b1 = *(const v8bf*)&lbc[(n0 + mrow) * 32 + 16 + khalf * 8];
      v16bf bfrag;
#pragma unroll
      for (int e = 0; e < 8; ++e) { bfrag[e] = b0[e]; bfrag[e + 8] = b1[e]; }
      acc[0][ni] = WMMA_BF16(afrag[0], bfrag, acc[0][ni]);
      acc[1][ni] = WMMA_BF16(afrag[1], bfrag, acc[1][ni]);
    }

    // ---- commit next buffer, single barrier per K-step ----
#ifdef ASYNC_LDS
    if (have_next) wait_async0();
#else
    if (have_next) {
      *(v8bf*)la_nxt       = ra0;
      *(v8bf*)(la_nxt + 8) = ra1;
#pragma unroll
      for (int kq = 0; kq < 4; ++kq) *(v8bf*)(lb_nxt + kq * 8) = rb[kq];
    }
#endif
    __syncthreads();
    cur = nxt;
  }

#pragma unroll
  for (int mi = 0; mi < 2; ++mi)
#pragma unroll
    for (int ni = 0; ni < 8; ++ni) {
      int gn = bn0 + wn * 128 + ni * 16 + mrow;
      float badd = bias ? bias[gn] : 0.0f;
#pragma unroll
      for (int r = 0; r < 8; ++r) {
        int gm = bm0 + wm * 32 + mi * 16 + khalf * 8 + r;
        float v = acc[mi][ni][r] + badd;
        if (act == 1) v = v > 0.0f ? v : 0.0f;
        else if (act == 2) v = v > 0.0f ? v : 0.01f * v;
        C[(size_t)gm * N + gn] = v;
      }
    }
}

// ---------------------------------------------------------------------------
// GCN scatter pipeline
// ---------------------------------------------------------------------------
__global__ void deg_init_kernel(float* __restrict__ deg, int n) {
  int i = blockIdx.x * blockDim.x + threadIdx.x;
  if (i < n) deg[i] = 1.0f;  // self loop
}
__global__ void deg_accum_kernel(const int* __restrict__ col, float* __restrict__ deg, int E) {
  int i = blockIdx.x * blockDim.x + threadIdx.x;
  if (i < E) atomicAdd(&deg[col[i]], 1.0f);
}
__global__ void dinv_kernel(float* __restrict__ deg, int n) {
  int i = blockIdx.x * blockDim.x + threadIdx.x;
  if (i < n) deg[i] = rsqrtf(deg[i]);  // deg >= 1 always
}
__global__ void agg_self_kernel(const float* __restrict__ h, const float* __restrict__ dinv,
                                float* __restrict__ agg) {
  int i = blockIdx.x, f = threadIdx.x;
  float s = dinv[i];
  agg[(size_t)i * 256 + f] = h[(size_t)i * 256 + f] * s * s;
}
__global__ void agg_edge_kernel(const int* __restrict__ row, const int* __restrict__ col,
                                const float* __restrict__ h, const float* __restrict__ dinv,
                                float* __restrict__ agg) {
  int e = blockIdx.x, f = threadIdx.x;
  int r = row[e], c = col[e];
  float nrm = dinv[r] * dinv[c];
  atomicAdd(&agg[(size_t)c * 256 + f], h[(size_t)r * 256 + f] * nrm);
}
__global__ void relu_pool_kernel(const float* __restrict__ agg, const float* __restrict__ bgcn,
                                 float* __restrict__ pooled) {
  int i = blockIdx.x, f = threadIdx.x;
  float v = fmaxf(agg[(size_t)i * 256 + f] + bgcn[f], 0.0f);
  int b = i >> 8;  // L = 256
  atomicMax((unsigned int*)&pooled[b * 256 + f], __float_as_uint(v));  // v >= 0
}

// ---------------------------------------------------------------------------
// LSTM recurrence on WMMA. grid = 8 batch-tiles of 16 rows, block = 128 (4 waves).
// Whh (n-major 512x128 bf16) resident in LDS (128 KB; CDNA5 WGP has 320 KB).
// ---------------------------------------------------------------------------
__device__ inline v16bf load_bfrag_lds(const bf16* p, int khalf) {
  v8bf b0 = *(const v8bf*)(p + khalf * 8);
  v8bf b1 = *(const v8bf*)(p + 16 + khalf * 8);
  v16bf r;
#pragma unroll
  for (int e = 0; e < 8; ++e) { r[e] = b0[e]; r[e + 8] = b1[e]; }
  return r;
}

__global__ __launch_bounds__(128) void lstm_rec_kernel(
    const float* __restrict__ xproj,  // (128, 256, 512)
    const bf16* __restrict__ whh,     // (512, 128) n-major bf16
    float* __restrict__ out, int col0, int Lsteps, int reverse, int only_t0) {
  extern __shared__ char smem[];
  bf16*  whh_lds = (bf16*)smem;                                    // 512*128
  bf16*  h_lds   = (bf16*)(smem + 512 * 128 * 2);                  // 16*128
  float* c_lds   = (float*)(smem + 512 * 128 * 2 + 16 * 128 * 2);  // 16*128

  const int t     = threadIdx.x;
  const int lane  = t & 31;
  const int w     = t >> 5;  // 0..3
  const int mrow  = lane & 15;
  const int khalf = lane >> 4;
  const int m0g   = blockIdx.x * 16;

  for (int i = t * 8; i < 512 * 128; i += 128 * 8)
    *(v8bf*)&whh_lds[i] = *(const v8bf*)&whh[i];
  for (int i = t; i < 16 * 128; i += 128) { h_lds[i] = (bf16)0.0f; c_lds[i] = 0.0f; }
  __syncthreads();

  for (int s = 0; s < Lsteps; ++s) {
    const int tt = reverse ? (255 - s) : s;
    const int tn = reverse ? tt - 1 : tt + 1;
    if (tn >= 0 && tn < 256)  // prefetch next step's gate rows
      __builtin_prefetch(&xproj[((size_t)(m0g + khalf * 8) * 256 + tn) * 512 + w * 16 + mrow], 0, 1);

    v16bf af[4];  // h rows for this wave's 16 batch rows, full K=128
#pragma unroll
    for (int k0 = 0; k0 < 4; ++k0) {
      v8bf a0 = *(const v8bf*)&h_lds[mrow * 128 + k0 * 32 + khalf * 8];
      v8bf a1 = *(const v8bf*)&h_lds[mrow * 128 + k0 * 32 + 16 + khalf * 8];
#pragma unroll
      for (int e = 0; e < 8; ++e) { af[k0][e] = a0[e]; af[k0][e + 8] = a1[e]; }
    }

    float hv[2][8];
#pragma unroll
    for (int jj = 0; jj < 2; ++jj) {
      const int jt = w + jj * 4;
      v8f gi, gf, gg, go;
#pragma unroll
      for (int e = 0; e < 8; ++e) { gi[e] = 0.f; gf[e] = 0.f; gg[e] = 0.f; go[e] = 0.f; }
#pragma unroll
      for (int k0 = 0; k0 < 4; ++k0) {
        const bf16* base = whh_lds + (size_t)(jt * 16 + mrow) * 128 + k0 * 32;
        v16bf bI = load_bfrag_lds(base, khalf);
        v16bf bF = load_bfrag_lds(base + 128 * 128, khalf);
        v16bf bG = load_bfrag_lds(base + 256 * 128, khalf);
        v16bf bO = load_bfrag_lds(base + 384 * 128, khalf);
        gi = WMMA_BF16(af[k0], bI, gi);
        gf = WMMA_BF16(af[k0], bF, gf);
        gg = WMMA_BF16(af[k0], bG, gg);
        go = WMMA_BF16(af[k0], bO, go);
      }
      const int j = jt * 16 + mrow;
#pragma unroll
      for (int r = 0; r < 8; ++r) {
        const int ml = khalf * 8 + r;
        const size_t xrow = ((size_t)(m0g + ml) * 256 + tt) * 512;
        float iv = gi[r] + xproj[xrow + j];
        float fv = gf[r] + xproj[xrow + 128 + j];
        float gv = gg[r] + xproj[xrow + 256 + j];
        float ov = go[r] + xproj[xrow + 384 + j];
        float is = 1.0f / (1.0f + __expf(-iv));
        float fs = 1.0f / (1.0f + __expf(-fv));
        float gt = tanhf(gv);
        float os = 1.0f / (1.0f + __expf(-ov));
        float cn = fs * c_lds[ml * 128 + j] + is * gt;
        c_lds[ml * 128 + j] = cn;
        hv[jj][r] = os * tanhf(cn);
      }
    }
    __syncthreads();  // all waves finished reading h_lds for this step
#pragma unroll
    for (int jj = 0; jj < 2; ++jj) {
      const int j = (w + jj * 4) * 16 + mrow;
#pragma unroll
      for (int r = 0; r < 8; ++r) {
        const int ml = khalf * 8 + r;
        h_lds[ml * 128 + j] = (bf16)hv[jj][r];
        if (!only_t0)
          out[((size_t)(m0g + ml) * 256 + tt) * 256 + col0 + j] = hv[jj][r];
        else if (tt == 0)
          out[(size_t)(m0g + ml) * 256 + col0 + j] = hv[jj][r];
      }
    }
    __syncthreads();
  }
}

// ---------------------------------------------------------------------------
// Head
// ---------------------------------------------------------------------------
__global__ void concat_bf_kernel(const float* __restrict__ pooled, const float* __restrict__ xseq,
                                 bf16* __restrict__ cat) {
  int m = blockIdx.x, f = threadIdx.x;  // block 512
  float v = (f < 256) ? pooled[m * 256 + f] : xseq[m * 256 + f - 256];
  cat[m * 512 + f] = (bf16)v;
}
__global__ void head_kernel(const float* __restrict__ out1, const float* __restrict__ W2,
                            const float* __restrict__ b2, float* __restrict__ out) {
  int m = blockIdx.x * blockDim.x + threadIdx.x;
  if (m >= 128) return;
  float z0 = b2[0], z1 = b2[1];
  for (int k = 0; k < 256; ++k) {
    float v = out1[m * 256 + k];
    z0 += v * W2[k * 2 + 0];
    z1 += v * W2[k * 2 + 1];
  }
  float mx = fmaxf(z0, z1);
  float lse = mx + logf(__expf(z0 - mx) + __expf(z1 - mx));
  out[m * 2 + 0] = z0 - lse;
  out[m * 2 + 1] = z1 - lse;
}

// ---------------------------------------------------------------------------
// Launch
// ---------------------------------------------------------------------------
extern "C" void kernel_launch(void* const* d_in, const int* in_sizes, int n_in,
                              void* d_out, int out_size, void* d_ws, size_t ws_size,
                              hipStream_t stream) {
  const float* x      = (const float*)d_in[0];
  const int*   ei     = (const int*)d_in[1];
  const float* W_gcn  = (const float*)d_in[3];
  const float* b_gcn  = (const float*)d_in[4];
  const float* l0_Wih = (const float*)d_in[5];
  const float* l0_Whh = (const float*)d_in[6];
  const float* l0_b   = (const float*)d_in[7];
  const float* l1_Wih = (const float*)d_in[8];
  const float* l1_Whh = (const float*)d_in[9];
  const float* l1_b   = (const float*)d_in[10];
  const float* W_seq  = (const float*)d_in[11];
  const float* b_seq  = (const float*)d_in[12];
  const float* W1     = (const float*)d_in[13];
  const float* b1     = (const float*)d_in[14];
  const float* W2     = (const float*)d_in[15];
  const float* b2     = (const float*)d_in[16];

  constexpr int N = 32768, Bn = 128, F = 768, NH = 256, H = 128, G4 = 512;
  const int E = in_sizes[1] / 2;

  char* ws = (char*)d_ws;
  size_t off = 0;
  auto alloc = [&](size_t bytes) -> void* {
    void* p = ws + off;
    off += (bytes + 255) & ~(size_t)255;
    return p;
  };
  bf16*  xbf    = (bf16*)alloc((size_t)N * F * 2);
  bf16*  wgcn   = (bf16*)alloc((size_t)NH * F * 2);       // n-major (256,768)
  bf16*  wih0   = (bf16*)alloc((size_t)2 * G4 * F * 2);   // already n-major
  bf16*  whh0   = (bf16*)alloc((size_t)2 * G4 * H * 2);
  bf16*  wih1   = (bf16*)alloc((size_t)2 * G4 * 2 * H * 2);
  bf16*  whh1   = (bf16*)alloc((size_t)2 * G4 * H * 2);
  bf16*  wseq   = (bf16*)alloc((size_t)NH * NH * 2);      // n-major
  bf16*  w1n    = (bf16*)alloc((size_t)NH * 2 * NH * 2);  // (256,512)
  float* hbuf   = (float*)alloc((size_t)N * NH * 4);
  float* agg    = (float*)alloc((size_t)N * NH * 4);
  float* deg    = (float*)alloc((size_t)N * 4);
  float* pooled = (float*)alloc((size_t)Bn * NH * 4);
  float* xproj0 = (float*)alloc((size_t)2 * N * G4 * 4);
  float* hs0    = (float*)alloc((size_t)N * 2 * H * 4);
  bf16*  hs0bf  = (bf16*)alloc((size_t)N * 2 * H * 2);
  float* xproj1 = (float*)alloc((size_t)2 * N * G4 * 4);
  float* h1out  = (float*)alloc((size_t)Bn * NH * 4);
  bf16*  h1bf   = (bf16*)alloc((size_t)Bn * NH * 2);
  float* xseq   = (float*)alloc((size_t)Bn * NH * 4);
  bf16*  catbf  = (bf16*)alloc((size_t)Bn * 2 * NH * 2);
  float* out1   = (float*)alloc((size_t)Bn * NH * 4);

  auto cdiv = [](size_t a, size_t b) { return (unsigned)((a + b - 1) / b); };
  const size_t SMEM = 512 * 128 * 2 + 16 * 128 * 2 + 16 * 128 * 4;  // 140 KB

  // conversions / weight layout
  f2bf_kernel<<<cdiv((size_t)N * F, 256), 256, 0, stream>>>(x, xbf, (size_t)N * F);
  transpose_f2bf_kernel<<<cdiv((size_t)F * NH, 256), 256, 0, stream>>>(W_gcn, wgcn, F, NH);
  f2bf_kernel<<<cdiv((size_t)2 * G4 * F, 256), 256, 0, stream>>>(l0_Wih, wih0, (size_t)2 * G4 * F);
  f2bf_kernel<<<cdiv((size_t)2 * G4 * H, 256), 256, 0, stream>>>(l0_Whh, whh0, (size_t)2 * G4 * H);
  f2bf_kernel<<<cdiv((size_t)2 * G4 * 2 * H, 256), 256, 0, stream>>>(l1_Wih, wih1, (size_t)2 * G4 * 2 * H);
  f2bf_kernel<<<cdiv((size_t)2 * G4 * H, 256), 256, 0, stream>>>(l1_Whh, whh1, (size_t)2 * G4 * H);
  transpose_f2bf_kernel<<<cdiv((size_t)NH * NH, 256), 256, 0, stream>>>(W_seq, wseq, NH, NH);
  transpose_f2bf_kernel<<<cdiv((size_t)2 * NH * NH, 256), 256, 0, stream>>>(W1, w1n, 2 * NH, NH);

  // GCN branch
  gemm_bf16_kernel<<<dim3(NH / 256, N / 128), 256, 0, stream>>>(xbf, wgcn, hbuf, N, NH, F, nullptr, 0);
  deg_init_kernel<<<cdiv(N, 256), 256, 0, stream>>>(deg, N);
  deg_accum_kernel<<<cdiv(E, 256), 256, 0, stream>>>(ei + E, deg, E);
  dinv_kernel<<<cdiv(N, 256), 256, 0, stream>>>(deg, N);
  agg_self_kernel<<<N, 256, 0, stream>>>(hbuf, deg, agg);
  agg_edge_kernel<<<E, 256, 0, stream>>>(ei, ei + E, hbuf, deg, agg);
  zero_f32_kernel<<<cdiv((size_t)Bn * NH, 256), 256, 0, stream>>>(pooled, (size_t)Bn * NH);
  relu_pool_kernel<<<N, 256, 0, stream>>>(agg, b_gcn, pooled);

  // LSTM layer 0 (input projections as big GEMMs, bias folded in)
  for (int d = 0; d < 2; ++d)
    gemm_bf16_kernel<<<dim3(G4 / 256, N / 128), 256, 0, stream>>>(
        xbf, wih0 + (size_t)d * G4 * F, xproj0 + (size_t)d * N * G4, N, G4, F, l0_b + d * G4, 0);
  lstm_rec_kernel<<<8, 128, SMEM, stream>>>(xproj0, whh0, hs0, 0, 256, 0, 0);
  lstm_rec_kernel<<<8, 128, SMEM, stream>>>(xproj0 + (size_t)N * G4, whh0 + G4 * H, hs0, 128, 256, 1, 0);
  f2bf_kernel<<<cdiv((size_t)N * 2 * H, 256), 256, 0, stream>>>(hs0, hs0bf, (size_t)N * 2 * H);

  // LSTM layer 1 (fwd needs only t=0 -> 1 step; bwd needs full 256 steps)
  for (int d = 0; d < 2; ++d)
    gemm_bf16_kernel<<<dim3(G4 / 256, N / 128), 256, 0, stream>>>(
        hs0bf, wih1 + (size_t)d * G4 * 2 * H, xproj1 + (size_t)d * N * G4, N, G4, 2 * H, l1_b + d * G4, 0);
  lstm_rec_kernel<<<8, 128, SMEM, stream>>>(xproj1, whh1, h1out, 0, 1, 0, 1);
  lstm_rec_kernel<<<8, 128, SMEM, stream>>>(xproj1 + (size_t)N * G4, whh1 + G4 * H, h1out, 128, 256, 1, 1);

  // Head
  f2bf_kernel<<<cdiv((size_t)Bn * NH, 256), 256, 0, stream>>>(h1out, h1bf, (size_t)Bn * NH);
  gemm_bf16_kernel<<<dim3(NH / 256, 1), 256, 0, stream>>>(h1bf, wseq, xseq, Bn, NH, NH, b_seq, 2);
  concat_bf_kernel<<<Bn, 512, 0, stream>>>(pooled, xseq, catbf);
  gemm_bf16_kernel<<<dim3(NH / 256, 1), 256, 0, stream>>>(catbf, w1n, out1, Bn, NH, 2 * NH, b1, 1);
  head_kernel<<<1, 128, 0, stream>>>(out1, W2, b2, (float*)d_out);
}
